// WaveletCenterNet_46420006535773
// MI455X (gfx1250) — matrix operand order repaired and checked by reading, not compiled, not run on hardware
//
#include <hip/hip_runtime.h>
#include <hip/hip_bf16.h>
#include <math.h>

typedef _Float16 h16;
typedef __attribute__((ext_vector_type(8)))  _Float16 v8h;
typedef __attribute__((ext_vector_type(16))) _Float16 v16h;
typedef __attribute__((ext_vector_type(8)))  float    v8f;
typedef __attribute__((ext_vector_type(4)))  int      v4i;

#define DEV static __device__ __forceinline__

#if defined(__has_builtin)
#if __has_builtin(__builtin_amdgcn_global_load_async_to_lds_b128)
#define HAS_ASYNC 1
#endif
#endif
#ifndef HAS_ASYNC
#define HAS_ASYNC 0
#endif

#define AS1 __attribute__((address_space(1)))
#define AS3 __attribute__((address_space(3)))

DEV void lds_stage16(const h16* src, h16* dst){
#if HAS_ASYNC
  __builtin_amdgcn_global_load_async_to_lds_b128(
      (AS1 v4i*)(unsigned long long)src,
      (AS3 v4i*)(unsigned long long)dst, 0, 0);
#else
  *(v8h*)dst = *(const v8h*)src;
#endif
}
DEV void wait_async0(){
#if HAS_ASYNC
#if __has_builtin(__builtin_amdgcn_s_wait_asynccnt)
  __builtin_amdgcn_s_wait_asynccnt(0);
#else
  asm volatile("s_wait_asynccnt 0" ::: "memory");
#endif
#endif
}

DEV v8f fzero8(){ v8f z;
#pragma unroll
  for(int i=0;i<8;++i) z[i]=0.f; return z; }

// A-operand fragment: p = &A[row*lda + k0] (row-major); lane half hf:
// elements 0..15 <-> K = hf*16 + e (contiguous)
DEV v16h frag_A(const h16* p, int hf){
  v8h lo = *(const v8h*)(p + hf*16);
  v8h hi = *(const v8h*)(p + hf*16 + 8);
  v16h r;
#pragma unroll
  for(int i=0;i<8;++i){ r[i]=lo[i]; r[i+8]=hi[i]; }
  return r;
}
// B-operand fragment: lane holds W[n, k0 + hf*16 .. +15] contiguous (32B load)
DEV v16h frag_B(const h16* p, int hf){ return *(const v16h*)(p + hf*16); }

DEV v8f wmma16(v16h a, v16h b, v8f c){
  return __builtin_amdgcn_wmma_f32_16x16x32_f16(false, a, false, b, (short)0, c, false, false);
}

DEV float gelu_f(float x){ return 0.5f*x*(1.0f+erff(x*0.70710678f)); }

// ---------------------------------------------------------------------------
// Workhorse GEMM: C[M,N] = act( (A[M,K] x W[N,K]^T) * s + b ) + R
// 4-wave block: 64(M) x 64(N) tile; A tile staged in LDS via async-to-LDS,
// shared by all 4 waves (each wave owns one 16-wide N tile).
// Requires M%64==0, K%32==0.  act: 0=none, 1=gelu, 2=clip(x/6+.5,0,1)
// ---------------------------------------------------------------------------
__global__ void k_gemm64(const h16* __restrict__ A, long lda,
                         const h16* __restrict__ W, int K, int N,
                         const float* __restrict__ sc, const float* __restrict__ bi,
                         const h16* __restrict__ R, long ldr,
                         h16* __restrict__ Ch, float* __restrict__ Cf, long ldc,
                         int act, int ntN){
  __shared__ __align__(32) h16 Atile[64*32];
  int tid  = threadIdx.x;            // 0..127
  int lane = tid & 31, wid = tid >> 5;
  int hf = lane >> 4, r = lane & 15;
  int ntG = (ntN + 3) >> 2;
  int tg  = blockIdx.x % ntG;
  long m0 = (long)(blockIdx.x / ntG) * 64;
  int n   = (tg*4 + wid)*16 + r;
  bool nv = (n < N);
  const h16* wrow = W + (long)(nv ? n : (N-1)) * K;   // clamped: no divergence in loop
  // cooperative A staging: 256 b128 chunks (64 rows x 4 octets), 2 per thread
  int row0 = tid >> 2, oct = tid & 3, row1 = row0 + 32;
  const h16* asrc0 = A + (m0 + row0)*lda + oct*8;
  const h16* asrc1 = A + (m0 + row1)*lda + oct*8;
  h16* adst0 = &Atile[row0*32 + oct*8];
  h16* adst1 = &Atile[row1*32 + oct*8];
  v8f cc[4];
#pragma unroll
  for(int i=0;i<4;++i) cc[i]=fzero8();
  for(int k0=0;k0<K;k0+=32){
    lds_stage16(asrc0 + k0, adst0);
    lds_stage16(asrc1 + k0, adst1);
    wait_async0();
    __syncthreads();
    v16h bf = frag_B(wrow + k0, hf);
    __builtin_prefetch(wrow + k0 + 32, 0, 3);
    v16h a0 = frag_A(Atile + (r     )*32, hf);
    v16h a1 = frag_A(Atile + (r + 16)*32, hf);
    v16h a2 = frag_A(Atile + (r + 32)*32, hf);
    v16h a3 = frag_A(Atile + (r + 48)*32, hf);
    cc[0]=wmma16(a0,bf,cc[0]); cc[1]=wmma16(a1,bf,cc[1]);
    cc[2]=wmma16(a2,bf,cc[2]); cc[3]=wmma16(a3,bf,cc[3]);
    __syncthreads();
  }
  if(!nv) return;
  float s_ = sc? sc[n] : 1.0f;
  float b_ = bi? bi[n] : 0.0f;
#pragma unroll
  for(int i=0;i<4;++i){
    long mb = m0 + i*16 + hf*8;
#pragma unroll
    for(int e=0;e<8;++e){
      float v = cc[i][e]*s_ + b_;
      if(act==1) v = gelu_f(v);
      else if(act==2) v = fminf(fmaxf(v*(1.0f/6.0f)+0.5f,0.f),1.f);
      long row = mb + e;
      if(R) v += (float)R[row*ldr + n];
      if(Cf) Cf[row*ldc + n] = v;
      else   Ch[row*ldc + n] = (h16)v;
    }
  }
}

// Attention logits: one wave = 16x16 tile of [Npad x Npad] per (b,h).
__global__ void k_attn_logits(const h16* __restrict__ Q, const h16* __restrict__ Kt,
                              const float* __restrict__ biases, const int* __restrict__ idxs,
                              h16* __restrict__ Lg, int N, int Npad, int C, int hd,
                              int heads, int nrel, float scale){
  int lane=threadIdx.x&31, hf=lane>>4, r=lane&15;
  int nt = Npad/16;
  int rest = blockIdx.x % (nt*nt);
  int bh   = blockIdx.x / (nt*nt);
  int tq = rest / nt, tk = rest % nt;
  int b = bh/heads, h = bh%heads;
  const h16* qrow = Q  + (((long)b*Npad + tq*16 + r)*C + h*hd);
  const h16* krow = Kt + (((long)b*Npad + tk*16 + r)*C + h*hd);
  v8f acc=fzero8();
  for(int k0=0;k0<hd;k0+=32){
    v16h a  = frag_A(qrow + k0, hf);
    v16h bb = frag_B(krow + k0, hf);
    acc = wmma16(a,bb,acc);
  }
  int mcol = tk*16 + r;
  h16* out = Lg + (long)bh*Npad*Npad;
#pragma unroll
  for(int e=0;e<8;++e){
    int nq = tq*16 + hf*8 + e;
    float v = acc[e]*scale;
    if(nq<N && mcol<N) v += biases[h*nrel + idxs[nq*N+mcol]];
    out[(long)nq*Npad + mcol] = (h16)v;
  }
}

// row softmax over valid N keys, zero the pad; one wave per row
__global__ void k_softmax(h16* __restrict__ Lg, int N, int Npad){
  long row = blockIdx.x;
  int n_g = (int)(row % Npad);
  h16* p = Lg + row*Npad;
  int lane = threadIdx.x;
  if(n_g >= N){ for(int m=lane;m<Npad;m+=32) p[m]=(h16)0; return; }
  float mx=-1e30f;
  for(int m=lane;m<N;m+=32) mx = fmaxf(mx,(float)p[m]);
  for(int o=16;o>0;o>>=1) mx = fmaxf(mx, __shfl_xor(mx,o,32));
  float sm=0.f;
  for(int m=lane;m<N;m+=32) sm += __expf((float)p[m]-mx);
  for(int o=16;o>0;o>>=1) sm += __shfl_xor(sm,o,32);
  float inv = 1.0f/sm;
  for(int m=lane;m<N;m+=32) p[m] = (h16)(__expf((float)p[m]-mx)*inv);
  for(int m=N+lane;m<Npad;m+=32) p[m]=(h16)0;
}

// AV: out = softmax(L) @ V^T per (b,h); replicates reference's (h,n,d)->(c,pos)
// reshape scramble; applies gelu; writes compact NHWC [B*N, C].
__global__ void k_attn_av(const h16* __restrict__ Aw, const h16* __restrict__ V,
                          h16* __restrict__ Oc, int N, int Npad, int C, int hd, int heads){
  int lane=threadIdx.x&31, hf=lane>>4, r=lane&15;
  int nt = Npad/16, dt = hd/16;
  int rest = blockIdx.x % (nt*dt);
  int bh   = blockIdx.x / (nt*dt);
  int tq = rest / dt, td = rest % dt;
  int b = bh/heads, h = bh%heads;
  const h16* arow = Aw + ((long)bh*Npad + tq*16 + r)*Npad;
  int d = td*16 + r;
  const h16* vcol = V + (long)b*Npad*C + h*hd + d;
  v8f acc=fzero8();
  for(int m0=0;m0<Npad;m0+=32){
    v16h a = frag_A(arow + m0, hf);
    v16h bb;
#pragma unroll
    for(int e=0;e<16;++e) bb[e] = vcol[(long)(m0 + hf*16 + e)*C];
    acc = wmma16(a,bb,acc);
  }
#pragma unroll
  for(int e=0;e<8;++e){
    int nq = tq*16 + hf*8 + e;
    if(nq < N){
      long f = ((long)h*N + nq)*hd + d;
      long cch = f / N, pos = f % N;
      Oc[((long)b*N + pos)*C + cch] = (h16)gelu_f(acc[e]);
    }
  }
}

// ---------------------- elementwise / spatial kernels ----------------------
__global__ void k_cvt(const float* __restrict__ s, h16* __restrict__ d, long n){
  long i = (long)blockIdx.x*blockDim.x + threadIdx.x;
  if(i<n) d[i]=(h16)s[i];
}
__global__ void k_nchw2nhwc(const float* __restrict__ X, h16* __restrict__ Y,
                            int C,int H,int W, long total){
  long i=(long)blockIdx.x*blockDim.x+threadIdx.x; if(i>=total) return;
  long t=i; int w=t%W; t/=W; int h=t%H; t/=H; int c=t%C; long b=t/C;
  Y[(((long)b*H+h)*W+w)*C + c] = (h16)X[i];
}
__global__ void k_conv3s2(const h16* __restrict__ X, const float* __restrict__ Wt,
                          const float* __restrict__ sc, const float* __restrict__ bi,
                          h16* __restrict__ Y, int Hi,int Wi,int Ci,int Ho,int Wo,int Co,
                          int dogelu, long total){
  long i=(long)blockIdx.x*blockDim.x+threadIdx.x; if(i>=total) return;
  long t=i; int co=t%Co; t/=Co; int ox=t%Wo; t/=Wo; int oy=t%Ho; long b=t/Ho;
  float acc=0.f;
  for(int ky=0;ky<3;++ky){ int iy=2*oy-1+ky; if(iy<0||iy>=Hi) continue;
    for(int kx=0;kx<3;++kx){ int ix=2*ox-1+kx; if(ix<0||ix>=Wi) continue;
      const h16* xp = X + (((long)b*Hi+iy)*Wi+ix)*Ci;
      for(int ci=0;ci<Ci;++ci)
        acc += (float)xp[ci]*Wt[(((long)co*Ci+ci)*3+ky)*3+kx];
    }}
  float v = acc*sc[co]+bi[co];
  if(dogelu) v = gelu_f(v);
  Y[i]=(h16)v;
}
__global__ void k_wavelet(const h16* __restrict__ X, h16* __restrict__ cA, h16* __restrict__ cL,
                          h16* __restrict__ cH, h16* __restrict__ cD,
                          int Hi,int Wi,int C,int Ho,int Wo, long total){
  long i=(long)blockIdx.x*blockDim.x+threadIdx.x; if(i>=total) return;
  long t=i; int c=t%C; t/=C; int ox=t%Wo; t/=Wo; int oy=t%Ho; long b=t/Ho;
  int y0=2*oy, x0=2*ox;
  const h16* xb = X + (long)b*Hi*Wi*C;
  float a  = (y0  <Hi && x0  <Wi)? (float)xb[((long)y0*Wi+x0)*C+c]     :0.f;
  float bb = (y0  <Hi && x0+1<Wi)? (float)xb[((long)y0*Wi+x0+1)*C+c]   :0.f;
  float cc = (y0+1<Hi && x0  <Wi)? (float)xb[((long)(y0+1)*Wi+x0)*C+c] :0.f;
  float dd = (y0+1<Hi && x0+1<Wi)? (float)xb[((long)(y0+1)*Wi+x0+1)*C+c]:0.f;
  const float s2 = 0.70710678f*0.70710678f;
  cA[i]=(h16)(s2*( a+bb+cc+dd));
  cL[i]=(h16)(s2*(-a+bb-cc+dd));
  cH[i]=(h16)(s2*(-a-bb+cc+dd));
  cD[i]=(h16)(s2*( a-bb-cc+dd));
}
__global__ void k_copy_to4(const h16* __restrict__ S, h16* __restrict__ D,
                           int C,int C4, long total){
  long i=(long)blockIdx.x*blockDim.x+threadIdx.x; if(i>=total) return;
  long m=i/C; int c=(int)(i%C);
  D[m*C4 + c]=S[i];
}
__global__ void k_repvgg(const h16* __restrict__ X, h16* __restrict__ Y,
                         const float* __restrict__ w3, const float* __restrict__ s3,
                         const float* __restrict__ b3, const float* __restrict__ w1,
                         const float* __restrict__ s1, const float* __restrict__ b1,
                         int H,int W,int C4, long total){
  long i=(long)blockIdx.x*blockDim.x+threadIdx.x; if(i>=total) return;
  long t=i; int c=t%C4; t/=C4; int x=t%W; t/=W; int y=t%H; long b=t/H;
  float acc=0.f;
#pragma unroll
  for(int ky=0;ky<3;++ky){ int iy=y-1+ky; if(iy<0||iy>=H) continue;
#pragma unroll
    for(int kx=0;kx<3;++kx){ int ix=x-1+kx; if(ix<0||ix>=W) continue;
      acc += (float)X[(((long)b*H+iy)*W+ix)*C4+c]*w3[c*9+ky*3+kx];
    }}
  float xc=(float)X[i];
  Y[i]=(h16)(acc*s3[c]+b3[c] + (xc*w1[c])*s1[c]+b1[c] + xc);
}
__global__ void k_extract(const h16* __restrict__ X4, h16* __restrict__ ll,
                          h16* __restrict__ high, int C, long total){
  long i=(long)blockIdx.x*blockDim.x+threadIdx.x; if(i>=total) return;
  long m=i/C; int c=(int)(i%C);
  const h16* rp = X4 + m*(long)(4*C);
  ll[i]=rp[c];
  high[i]=(h16)(((float)rp[C+c]+(float)rp[2*C+c]+(float)rp[3*C+c])*(1.0f/3.0f));
}
__global__ void k_rowmean(const h16* __restrict__ X, h16* __restrict__ out, int N, int C){
  int b=blockIdx.x;
  for(int c=threadIdx.x;c<C;c+=blockDim.x){
    float s=0.f;
    for(int n=0;n<N;++n) s += (float)X[((long)b*N+n)*C + c];
    out[(long)b*C+c]=(h16)(s/(float)N);
  }
}
__global__ void k_scale(const h16* __restrict__ X, const h16* __restrict__ G,
                        h16* __restrict__ Y, int N,int C, long total){
  long i=(long)blockIdx.x*blockDim.x+threadIdx.x; if(i>=total) return;
  int c=(int)(i%C); long b=i/((long)N*C);
  Y[i]=(h16)((float)X[i]*(float)G[b*C+c]);
}
__global__ void k_pad(const h16* __restrict__ S, h16* __restrict__ D,
                      int N,int Np,int C, long total){
  long i=(long)blockIdx.x*blockDim.x+threadIdx.x; if(i>=total) return;
  int c=(int)(i%C); long t=i/C; int n=(int)(t%Np); long b=t/Np;
  D[i] = (n<N)? S[((long)b*N+n)*C+c] : (h16)0;
}

// ---------------------------------------------------------------------------
static inline int gsz(long tot){ return (int)((tot+255)/256); }

static void gemm(hipStream_t st, const h16* A, long lda, const h16* W, int K, int N,
                 const float* sc, const float* bi, const h16* R, long ldr,
                 h16* Ch, float* Cf, long ldc, int M, int act){
  int ntN=(N+15)/16, ntM=M/64, ntG=(ntN+3)/4;
  k_gemm64<<<ntM*ntG,128,0,st>>>(A,lda,W,K,N,sc,bi,R,ldr,Ch,Cf,ldc,act,ntN);
}

extern "C" void kernel_launch(void* const* d_in, const int* in_sizes, int n_in,
                              void* d_out, int out_size, void* d_ws, size_t ws_size,
                              hipStream_t stream){
  (void)in_sizes;(void)n_in;(void)out_size;(void)ws_size;
  const int EMB[3]={128,256,384}, DEP[3]={2,3,3}, RESA[3]={14,7,4};
  const int Bn=64, heads=4;

  // --- input walk (jax pytree flatten order: dict keys sorted) ---
  int cur=0;
  auto F=[&]()->const float*{ return (const float*)d_in[cur++]; };
  const int* bidx[3]; for(int i=0;i<3;++i) bidx[i]=(const int*)d_in[cur++];
  const float* headWf=F(); const float* headb=F();

  char* base=(char*)d_ws; size_t off=0;
  auto ah=[&](size_t n)->h16*{ h16* p=(h16*)(base+off); off=(off+n*2+255)&~(size_t)255; return p; };
  auto cvt=[&](const float* s, size_t n)->h16*{
    h16* d=ah(n); k_cvt<<<gsz((long)n),256,0,stream>>>(s,d,(long)n); return d; };

  h16* headW = cvt(headWf,(size_t)1000*384);

  struct BW { const float* biases; h16* projW; const float *projb,*projs;
    h16* fm1W; const float *fm1b,*fm1s; h16* fm2W; const float *fm2b,*fm2s;
    h16* heseW; const float* heseb; h16* hm1W; const float *hm1b,*hm1s;
    h16* hm2W; const float *hm2b,*hm2s; h16* leseW; const float* leseb;
    const float *r1W,*r1b,*r1s,*r3W,*r3b,*r3s; };
  struct SW { BW bw[3]; h16 *wll,*wlh,*whl,*whh;
    const float *llb,*lls,*lhb,*lhs,*hlb,*hls,*hhb,*hhs; };
  SW st[3];
  int cin=128;
  for(int i=0;i<3;++i){
    int c=EMB[i];
    for(int k=0;k<DEP[i];++k){
      BW& b=st[i].bw[k];
      b.biases=F();
      b.projW=cvt(F(),(size_t)c*c);     b.projb=F(); b.projs=F();
      b.fm1W =cvt(F(),(size_t)2*c*c);   b.fm1b=F();  b.fm1s=F();
      b.fm2W =cvt(F(),(size_t)c*2*c);   b.fm2b=F();  b.fm2s=F();
      b.heseW=cvt(F(),(size_t)c*c);     b.heseb=F();
      b.hm1W =cvt(F(),(size_t)2*c*c);   b.hm1b=F();  b.hm1s=F();
      b.hm2W =cvt(F(),(size_t)c*2*c);   b.hm2b=F();  b.hm2s=F();
      b.leseW=cvt(F(),(size_t)c*c);     b.leseb=F();
      b.r1W=F(); b.r1b=F(); b.r1s=F();
      b.r3W=F(); b.r3b=F(); b.r3s=F();
    }
    st[i].whh=cvt(F(),(size_t)c*cin); st[i].hhb=F(); st[i].hhs=F();
    st[i].whl=cvt(F(),(size_t)c*cin); st[i].hlb=F(); st[i].hls=F();
    st[i].wlh=cvt(F(),(size_t)c*cin); st[i].lhb=F(); st[i].lhs=F();
    st[i].wll=cvt(F(),(size_t)c*cin); st[i].llb=F(); st[i].lls=F();
    cin=c;
  }
  const float *s0W=F(),*s0b=F(),*s0s=F();
  const float *s1W=F(),*s1b=F(),*s1s=F();
  const float *s2W=F(),*s2b=F(),*s2s=F();
  const float* x=F();

  h16* ll28 = ah((size_t)Bn*28*28*128);          // stem output, stage0 input
  size_t phase0 = off;

  // ---------------- stem (phase A buffers) ----------------
  {
    h16* xh = ah((size_t)Bn*224*224*3);
    h16* sA = ah((size_t)Bn*112*112*32);
    h16* sB = ah((size_t)Bn*56*56*64);
    long tot=(long)Bn*3*224*224;
    k_nchw2nhwc<<<gsz(tot),256,0,stream>>>(x,xh,3,224,224,tot);
    tot=(long)Bn*112*112*32;
    k_conv3s2<<<gsz(tot),256,0,stream>>>(xh,s0W,s0s,s0b,sA,224,224,3,112,112,32,1,tot);
    tot=(long)Bn*56*56*64;
    k_conv3s2<<<gsz(tot),256,0,stream>>>(sA,s1W,s1s,s1b,sB,112,112,32,56,56,64,1,tot);
    tot=(long)Bn*28*28*128;
    k_conv3s2<<<gsz(tot),256,0,stream>>>(sB,s2W,s2s,s2b,ll28,56,56,64,28,28,128,0,tot);
  }

  // ---------------- stage arena (phase B, reuses stem region) ----------------
  off = phase0;
  h16* X4  = ah(6422528);  h16* X4b = ah(6422528);
  h16* cA  = ah(1605632);  h16* cL  = ah(1605632);
  h16* cH  = ah(1605632);  h16* cD  = ah(1605632);
  h16* LR  = ah(1605632);  h16* lsA = ah(1605632); h16* lsB = ah(1605632);
  h16* hig = ah(1605632);  h16* seh = ah(1605632);
  h16* qc  = ah(1605632);  h16* kc  = ah(1605632); h16* oc  = ah(1605632);
  h16* tmp2= ah(3211264);
  h16* mea = ah(64*1536);  h16* gat = ah(64*1536);
  h16* Qp  = ah(1835008);  h16* Kp  = ah(1835008); h16* Vp  = ah(1835008);
  h16* Lg  = ah(12845056);

  h16* llin = ll28; int Hi=28; cin=128;
  for(int i=0;i<3;++i){
    int c=EMB[i], Ho=RESA[i], N=Ho*Ho, M=Bn*N, Np=((N+31)/32)*32;
    int hd=c/heads, nrel=N;
    float scale=1.0f/sqrtf((float)hd);
    long tot=(long)Bn*Ho*Ho*cin;
    k_wavelet<<<gsz(tot),256,0,stream>>>(llin,cA,cL,cH,cD,Hi,Hi,cin,Ho,Ho,tot);
    gemm(stream,cA,cin,st[i].wll,cin,c,st[i].lls,st[i].llb,nullptr,0,lsA ,nullptr,c  ,M,0);
    gemm(stream,cL,cin,st[i].wlh,cin,c,st[i].lhs,st[i].lhb,nullptr,0,X4+c ,nullptr,4*c,M,0);
    gemm(stream,cH,cin,st[i].whl,cin,c,st[i].hls,st[i].hlb,nullptr,0,X4+2*c,nullptr,4*c,M,0);
    gemm(stream,cD,cin,st[i].whh,cin,c,st[i].hhs,st[i].hhb,nullptr,0,X4+3*c,nullptr,4*c,M,0);
    h16* ll=lsA;
    for(int k=0;k<DEP[i];++k){
      BW& bw=st[i].bw[k];
      long MC=(long)M*c, M4=(long)M*4*c, PC=(long)Bn*Np*c;
      k_copy_to4<<<gsz(MC),256,0,stream>>>(ll,X4,c,4*c,MC);
      k_repvgg<<<gsz(M4),256,0,stream>>>(X4,X4b,bw.r3W,bw.r3s,bw.r3b,bw.r1W,bw.r1s,bw.r1b,Ho,Ho,4*c,M4);
      k_extract<<<gsz(MC),256,0,stream>>>(X4b,LR,hig,c,MC);
      // q = ffn(se(high, hese), hmlp)
      k_rowmean<<<Bn,256,0,stream>>>(hig,mea,N,c);
      gemm(stream,mea,c,bw.heseW,c,c,nullptr,bw.heseb,nullptr,0,gat,nullptr,c,Bn,2);
      k_scale<<<gsz(MC),256,0,stream>>>(hig,gat,seh,N,c,MC);
      gemm(stream,seh ,c  ,bw.hm1W,c  ,2*c,bw.hm1s,bw.hm1b,nullptr,0,tmp2,nullptr,2*c,M,1);
      gemm(stream,tmp2,2*c,bw.hm2W,2*c,c  ,bw.hm2s,bw.hm2b,nullptr,0,qc  ,nullptr,c  ,M,0);
      // k = se(ll, lese)
      k_rowmean<<<Bn,256,0,stream>>>(LR,mea,N,c);
      gemm(stream,mea,c,bw.leseW,c,c,nullptr,bw.leseb,nullptr,0,gat,nullptr,c,Bn,2);
      k_scale<<<gsz(MC),256,0,stream>>>(LR,gat,kc,N,c,MC);
      // attention (padded token domain)
      k_pad<<<gsz(PC),256,0,stream>>>(qc,Qp,N,Np,c,PC);
      k_pad<<<gsz(PC),256,0,stream>>>(kc,Kp,N,Np,c,PC);
      k_pad<<<gsz(PC),256,0,stream>>>(LR,Vp,N,Np,c,PC);
      int nt=Np/16, dt=hd/16;
      k_attn_logits<<<Bn*heads*nt*nt,32,0,stream>>>(Qp,Kp,bw.biases,bidx[i],Lg,N,Np,c,hd,heads,nrel,scale);
      k_softmax<<<Bn*heads*Np,32,0,stream>>>(Lg,N,Np);
      k_attn_av<<<Bn*heads*nt*dt,32,0,stream>>>(Lg,Vp,oc,N,Np,c,hd,heads);
      // ll = ll + proj(attn_out);  ll = ll + ffn(ll, fmlp)
      gemm(stream,oc  ,c  ,bw.projW,c  ,c  ,bw.projs,bw.projb,LR ,c,lsA,nullptr,c  ,M,0);
      gemm(stream,lsA ,c  ,bw.fm1W ,c  ,2*c,bw.fm1s ,bw.fm1b ,nullptr,0,tmp2,nullptr,2*c,M,1);
      gemm(stream,tmp2,2*c,bw.fm2W ,2*c,c  ,bw.fm2s ,bw.fm2b ,lsA,c,lsB,nullptr,c  ,M,0);
      ll=lsB;
    }
    llin=ll; Hi=Ho; cin=c;
  }

  // head: pooled @ W^T + b  (fp32 output)
  k_rowmean<<<Bn,256,0,stream>>>(llin,mea,16,384);
  gemm(stream,mea,384,headW,384,1000,nullptr,headb,nullptr,0,nullptr,(float*)d_out,1000,Bn,0);
}
